// BidirectionalRNN_3770981286128
// MI455X (gfx1250) — matrix operand order repaired and checked by reading, not compiled, not run on hardware
//
#include <hip/hip_runtime.h>
#include <hip/hip_bf16.h>

// ---------------------------------------------------------------------------
// Bidirectional LSTM, B=16, S=2048, D=H=512.
// Persistent kernel, bf16 WMMA (v_wmma_f32_16x16x32_bf16), fused
// z = [x_t ; h] @ [W;U] (K=1024) per timestep, grid barrier between steps.
// Async global->LDS staging (ASYNCcnt) when the toolchain exposes it.
// ---------------------------------------------------------------------------

typedef __attribute__((ext_vector_type(16))) __bf16 v16bf;
typedef __attribute__((ext_vector_type(8)))  float  v8f;
typedef __attribute__((ext_vector_type(4)))  int    v4i;

#define BB   16      // batch
#define SS   2048    // seq len
#define DD   512     // input dim
#define HH   512     // hidden dim
#define G4H  2048    // 4*H
#define NBLK 16      // persistent blocks: 2 dirs x 8 column slices
#define WGCOLS 64    // h-columns owned per block (512/8)

// ---- async global->LDS support probe ----
#if defined(__has_builtin)
#  if __has_builtin(__builtin_amdgcn_global_load_async_to_lds_b128) && \
      __has_builtin(__builtin_amdgcn_s_wait_asynccnt)
#    define HAVE_ASYNC_LDS 1
#  endif
#endif
#ifndef HAVE_ASYNC_LDS
#  define HAVE_ASYNC_LDS 0
#endif

// ---- workspace layout (bytes) ----
// [0, 8 MiB)        : WU pack  bf16  [2 dirs][128 ntile][32 ktile][32 lane][16]
// [8 MiB, 40 MiB)   : xA pack  bf16  [S][16 ktile][32 lane][16]
// [40 MiB, +64 KiB) : hA pack  bf16  [2 dirs][2 buf][16 ktile][32 lane][16]
// then              : barrier  u32[2]
#define WS_WU_OFF   0ull
#define WS_XA_OFF   (8ull  << 20)
#define WS_HA_OFF   (40ull << 20)
#define WS_BAR_OFF  ((40ull << 20) + 65536ull)

#define OUT_STATE_BASE ((size_t)BB * SS * 2 * HH)   // 33,554,432 floats

__device__ __forceinline__ unsigned short f32_to_bf16_rne(float f) {
  unsigned u = __float_as_uint(f);
  u += 0x7FFFu + ((u >> 16) & 1u);           // round-to-nearest-even
  return (unsigned short)(u >> 16);
}

__device__ __forceinline__ float sigmoidf_fast(float x) {
  return 1.0f / (1.0f + __expf(-x));
}

// ---------------------------------------------------------------------------
// Init: zero hA double buffers + barrier counters (d_ws is poisoned 0xAA).
// ---------------------------------------------------------------------------
__global__ void lstm_init_kernel(unsigned short* hA, unsigned* barrier) {
  int gid = blockIdx.x * blockDim.x + threadIdx.x;
  if (gid < 2 * 2 * 8192) hA[gid] = 0;
  if (gid < 2)            barrier[gid] = 0;
}

// ---------------------------------------------------------------------------
// Pack W||U (f32 row-major [K,4H]) into B-matrix bf16 lane layout.
// B 32x16 tile layout (16-bit): lane = n + 16*((kk>>3)&1), elem = (kk&7) + 8*(kk>>4).
// ---------------------------------------------------------------------------
__global__ void pack_wu_kernel(const float* __restrict__ W_fw,
                               const float* __restrict__ U_fw,
                               const float* __restrict__ W_bw,
                               const float* __restrict__ U_bw,
                               unsigned short* __restrict__ wupack) {
  unsigned o = blockIdx.x * 256u + threadIdx.x;   // 2 * 2^21 total
  if (o >= 2u * 128u * 32u * 32u * 16u) return;
  int elem = o & 15;
  int lane = (o >> 4) & 31;
  int kt   = (o >> 9) & 31;
  int nt   = (o >> 14) & 127;
  int dir  = (int)(o >> 21);
  int n    = lane & 15;
  int kk   = (elem & 7) + 8 * (lane >> 4) + 16 * (elem >> 3);
  int col  = nt * 16 + n;
  int krow = kt * 32 + kk;                        // 0..1023 (W rows then U rows)
  const float* W = dir ? W_bw : W_fw;
  const float* U = dir ? U_bw : U_fw;
  float v = (krow < DD) ? W[(size_t)krow * G4H + col]
                        : U[(size_t)(krow - DD) * G4H + col];
  wupack[o] = f32_to_bf16_rne(v);
}

// ---------------------------------------------------------------------------
// Pack x ([B,S,D] f32) into A-matrix bf16 lane layout per timestep.
// A 16x32 tile layout (16-bit): lane = m + 16*((kk>>3)&1),
//                               elem = (kk&7) + 8*(kk>>4).
// ---------------------------------------------------------------------------
__global__ void pack_x_kernel(const float* __restrict__ x,
                              unsigned short* __restrict__ xapack) {
  unsigned o = blockIdx.x * 256u + threadIdx.x;   // 2^24 total
  if (o >= (unsigned)SS * 16u * 32u * 16u) return;
  int elem = o & 15;
  int lane = (o >> 4) & 31;
  int kt   = (o >> 9) & 15;
  int s    = (int)(o >> 13);
  int m    = lane & 15;
  int kk   = (elem & 7) | ((lane >> 4) << 3) | ((elem >> 3) << 4);
  int d    = kt * 32 + kk;
  xapack[o] = f32_to_bf16_rne(x[(size_t)m * SS * DD + (size_t)s * DD + d]);
}

// ---------------------------------------------------------------------------
// Persistent recurrence kernel. 16 blocks x 512 threads (16 waves).
// block b: dir = b/8, wg = b%8 owns h-columns [wg*64, wg*64+64).
// Per step: 16 waves x 32 chained v_wmma_f32_16x16x32_bf16 (K=1024 fused).
// ---------------------------------------------------------------------------
__global__ __launch_bounds__(512, 1)
void lstm_persistent_kernel(const unsigned short* __restrict__ wupack,
                            const unsigned short* __restrict__ xapack,
                            const float* __restrict__ b_fw,
                            const float* __restrict__ b_bw,
                            unsigned short* __restrict__ hA,
                            unsigned* __restrict__ barrier,
                            float* __restrict__ out) {
  __shared__ __align__(16) unsigned short sXA[8192];   // x_t A-pack (16 KiB)
  __shared__ __align__(16) unsigned short sHA[8192];   // h   A-pack (16 KiB)
  __shared__ float sZ[4][16][WGCOLS];                  // gate pre-acts (16 KiB)

  const int dir  = blockIdx.x >> 3;
  const int wg   = blockIdx.x & 7;
  const int tid  = threadIdx.x;
  const int wave = tid >> 5;
  const int lane = tid & 31;
  const float* bias = dir ? b_bw : b_fw;

  // Fixed phase-C element mapping: thread tid owns elements tid and tid+512.
  // Keep bias slices and cell state in registers for the whole recurrence.
  float bReg[2][4];
  float cReg[2] = {0.0f, 0.0f};
  #pragma unroll
  for (int j = 0; j < 2; ++j) {
    const int e = tid + j * 512;
    const int col = wg * WGCOLS + (e & 63);
    bReg[j][0] = bias[0 * HH + col];
    bReg[j][1] = bias[1 * HH + col];
    bReg[j][2] = bias[2 * HH + col];
    bReg[j][3] = bias[3 * HH + col];
  }

  // this wave's 16-column z tile: gate g, sub-slice within gate
  const int g     = wave >> 2;
  const int sub   = wave & 3;
  const int ntile = g * 32 + wg * 4 + sub;            // 0..127
  const unsigned short* bbase =
      wupack + ((size_t)(dir * 128 + ntile)) * 16384;  // 32 kt * 32 lane * 16

  for (int s = 0; s < SS; ++s) {
    const int t   = dir ? (SS - 1 - s) : s;
    const int cur = s & 1;
    const int nxt = cur ^ 1;

    // ---- Phase A: stage x_t and h A-packs into LDS (32 KiB total) ----
    {
      const unsigned short* gx = xapack + (size_t)t * 8192;
      const unsigned short* gh = hA + (size_t)(dir * 2 + cur) * 8192;
#if HAVE_ASYNC_LDS
      for (int i = tid; i < 1024; i += 512) {          // 1024 x 16B per buffer
        __builtin_amdgcn_global_load_async_to_lds_b128(
            (v4i*)(gx + (size_t)i * 8), (v4i*)(&sXA[i * 8]), 0, 0);
        __builtin_amdgcn_global_load_async_to_lds_b128(
            (v4i*)(gh + (size_t)i * 8), (v4i*)(&sHA[i * 8]), 0, 0);
      }
      __builtin_amdgcn_s_wait_asynccnt(0);
#else
      const uint4* gx4 = (const uint4*)gx;
      const uint4* gh4 = (const uint4*)gh;
      uint4* sx = (uint4*)sXA;
      uint4* sh = (uint4*)sHA;
      for (int i = tid; i < 1024; i += 512) { sx[i] = gx4[i]; sh[i] = gh4[i]; }
#endif
    }
    __syncthreads();

    // ---- Phase B: z tile = [x_t ; h] @ [W;U] slice, K=1024, 32 WMMAs ----
    v8f acc = {};
    #pragma unroll 4
    for (int kt = 0; kt < 16; ++kt) {                  // x half (W rows)
      v16bf a = *(const v16bf*)(&sXA[kt * 512 + lane * 16]);
      v16bf b = *(const v16bf*)(bbase + (size_t)kt * 512 + lane * 16);
      acc = __builtin_amdgcn_wmma_f32_16x16x32_bf16(
          false, a, false, b, (short)0, acc, false, false);
    }
    #pragma unroll 4
    for (int kt = 0; kt < 16; ++kt) {                  // h half (U rows)
      v16bf a = *(const v16bf*)(&sHA[kt * 512 + lane * 16]);
      v16bf b = *(const v16bf*)(bbase + (size_t)(16 + kt) * 512 + lane * 16);
      acc = __builtin_amdgcn_wmma_f32_16x16x32_bf16(
          false, a, false, b, (short)0, acc, false, false);
    }
    // C/D layout: VGPR r -> M = r + 8*(lane>>4), N = lane&15
    {
      const int n = lane & 15;
      const int mh = (lane >> 4) * 8;
      #pragma unroll
      for (int r = 0; r < 8; ++r)
        sZ[g][mh + r][sub * 16 + n] = acc[r];
    }
    __syncthreads();

    // ---- Phase C: gate math, update c/h, emit outputs, repack h ----
    // Prefetch next timestep's x A-pack while doing scalar-heavy gate math.
    if (s < SS - 1 && tid < 128) {
      const int tn = dir ? (t - 1) : (t + 1);
      __builtin_prefetch(xapack + (size_t)tn * 8192 + (size_t)tid * 64, 0, 0);
    }
    #pragma unroll
    for (int j = 0; j < 2; ++j) {
      const int e  = tid + j * 512;
      const int m  = e >> 6;
      const int hc = e & 63;
      const int col = wg * WGCOLS + hc;                // 0..511 hidden column
      float zi = sZ[0][m][hc] + bReg[j][0];
      float zf = sZ[1][m][hc] + bReg[j][1];
      float zg = sZ[2][m][hc] + bReg[j][2];
      float zo = sZ[3][m][hc] + bReg[j][3];
      float cn = sigmoidf_fast(zf) * cReg[j] + sigmoidf_fast(zi) * tanhf(zg);
      cReg[j] = cn;
      float h = sigmoidf_fast(zo) * tanhf(cn);

      // sequence output [B,S,2H]
      out[((size_t)m * SS + t) * (2 * HH) + (size_t)dir * HH + col] = h;

      // repack h into A-matrix bf16 layout for next step
      const int kt = col >> 5, kk = col & 31;
      const int la = m + 16 * ((kk >> 3) & 1);
      const int el = (kk & 7) + 8 * (kk >> 4);
      hA[((size_t)(dir * 2 + nxt) * 16 + kt) * 512 + la * 16 + el] =
          f32_to_bf16_rne(h);

      if (s == SS - 1) {                               // final h, c states
        const size_t hb = OUT_STATE_BASE + (size_t)dir * 2 * BB * HH;
        out[hb + (size_t)m * HH + col] = h;
        out[hb + (size_t)BB * HH + (size_t)m * HH + col] = cn;
      }
    }

    // ---- Phase D: device-wide barrier (all 16 blocks advance together) ----
    __syncthreads();
    if (tid == 0) {
      __threadfence();
      const unsigned want = (unsigned)(s + 1);
      unsigned arrived = atomicAdd(&barrier[0], 1u) + 1u;
      if (arrived == want * (unsigned)NBLK) {
        __threadfence();
        atomicAdd(&barrier[1], 1u);
      }
      while (__hip_atomic_load(&barrier[1], __ATOMIC_ACQUIRE,
                               __HIP_MEMORY_SCOPE_AGENT) < want) {
        __builtin_amdgcn_s_sleep(2);
      }
      __threadfence();
    }
    __syncthreads();
  }
}

// ---------------------------------------------------------------------------
extern "C" void kernel_launch(void* const* d_in, const int* in_sizes, int n_in,
                              void* d_out, int out_size, void* d_ws, size_t ws_size,
                              hipStream_t stream) {
  const float* x    = (const float*)d_in[0];
  const float* W_fw = (const float*)d_in[1];
  const float* U_fw = (const float*)d_in[2];
  const float* b_fw = (const float*)d_in[3];
  const float* W_bw = (const float*)d_in[4];
  const float* U_bw = (const float*)d_in[5];
  const float* b_bw = (const float*)d_in[6];
  float* out = (float*)d_out;

  char* ws = (char*)d_ws;
  unsigned short* wupack = (unsigned short*)(ws + WS_WU_OFF);
  unsigned short* xapack = (unsigned short*)(ws + WS_XA_OFF);
  unsigned short* hA     = (unsigned short*)(ws + WS_HA_OFF);
  unsigned*       bar    = (unsigned*)(ws + WS_BAR_OFF);

  // 1) zero hA double buffers + barrier counters
  lstm_init_kernel<<<(2 * 2 * 8192 + 255) / 256, 256, 0, stream>>>(hA, bar);

  // 2) pack W||U into bf16 B-matrix layout (2 * 2^21 elements)
  pack_wu_kernel<<<(2u << 21) / 256, 256, 0, stream>>>(W_fw, U_fw, W_bw, U_bw,
                                                       wupack);

  // 3) pack x into bf16 A-matrix layout (2^24 elements)
  pack_x_kernel<<<(1u << 24) / 256, 256, 0, stream>>>(x, xapack);

  // 4) persistent bidirectional recurrence: 16 blocks x 512 threads
  lstm_persistent_kernel<<<NBLK, 512, 0, stream>>>(wupack, xapack, b_fw, b_bw,
                                                   hA, bar, out);
}
